// Policy_54150947668254
// MI455X (gfx1250) — matrix operand order repaired and checked by reading, not compiled, not run on hardware
//
#include <hip/hip_runtime.h>

typedef _Float16 h16;
typedef __attribute__((ext_vector_type(16))) _Float16 v16h;
typedef __attribute__((ext_vector_type(8)))  float    v8f;

#define L_   25
#define W_   11
#define H_   11
#define BOX  3025   // 25*11*11
#define M_   200
#define AP_  112    // actions padded to 7*16

// Branch-free tanh: sign(x) * (1 - e^{-2|x|}) / (1 + e^{-2|x|}).
// Uses native v_exp_f32 / v_rcp_f32; saturates to +/-1 for large |x| since
// exp2 underflows to 0. No EXEC divergence (WMMA loops must keep EXEC all-1s).
__device__ __forceinline__ float fast_tanh(float x) {
  float ax = __builtin_fabsf(x);
  float y  = __builtin_amdgcn_exp2f(-2.885390081777927f * ax);  // e^{-2|x|}
  float t  = (1.0f - y) * __builtin_amdgcn_rcpf(1.0f + y);
  return __builtin_copysignf(t, x);
}

// ---------------- prep kernels (one-time per launch, tiny) ----------------

// G = enc_w(128x128) @ fc1_w(128x64)  -> [128,64]
__global__ void pk1_G(const float* __restrict__ enc_w, const float* __restrict__ fc1_w,
                      float* __restrict__ G) {
  int t = blockIdx.x * blockDim.x + threadIdx.x;
  if (t >= 128 * 64) return;
  int hh = t >> 6, o = t & 63;
  float s = 0.f;
  for (int m = 0; m < 128; ++m) s += enc_w[hh * 128 + m] * fc1_w[m * 64 + o];
  G[hh * 64 + o] = s;
}

// K2 = G @ C2 where C2[o, c*9+i*3+j] = cnn2_w[o,c,i,j]  -> [128,576]
__global__ void pk2_K2(const float* __restrict__ G, const float* __restrict__ w2,
                       float* __restrict__ K2) {
  int t = blockIdx.x * blockDim.x + threadIdx.x;
  if (t >= 128 * 576) return;
  int hh = t / 576, c9 = t % 576;
  float s = 0.f;
  for (int o = 0; o < 64; ++o) s += G[hh * 64 + o] * w2[o * 576 + c9];
  K2[hh * 576 + c9] = s;
}

// T[idx=ch*121+x*11+y][h] = sum_{oc,px,py valid} K2[h, oc*9+px*3+py] * w1[oc,ch,x-3px,y-3py]
__global__ void pk3_T(const float* __restrict__ K2, const float* __restrict__ w1,
                      float* __restrict__ T) {
  int t = blockIdx.x * blockDim.x + threadIdx.x;
  if (t >= BOX * 128) return;
  int idx = t >> 7, hh = t & 127;
  int ch = idx / 121, rem = idx % 121, x = rem / 11, y = rem % 11;
  float s = 0.f;
  for (int px = 0; px < 3; ++px) {
    int i = x - 3 * px; if (i < 0 || i > 4) continue;
    for (int py = 0; py < 3; ++py) {
      int j = y - 3 * py; if (j < 0 || j > 4) continue;
      for (int oc = 0; oc < 64; ++oc)
        s += K2[hh * 576 + oc * 9 + px * 3 + py] * w1[oc * 625 + ch * 25 + i * 5 + j];
    }
  }
  T[idx * 128 + hh] = s;
}

// bias_h = K2 @ b1_broadcast + G @ b2 + enc_w @ f1b + eb
__global__ void pk4_bias(const float* __restrict__ K2, const float* __restrict__ G,
                         const float* __restrict__ enc_w, const float* __restrict__ b1,
                         const float* __restrict__ b2, const float* __restrict__ f1b,
                         const float* __restrict__ eb, float* __restrict__ bias_h) {
  int hh = threadIdx.x;
  if (hh >= 128) return;
  float s = eb[hh];
  for (int oc = 0; oc < 64; ++oc) {
    float bb = b1[oc];
    for (int p = 0; p < 9; ++p) s += K2[hh * 576 + oc * 9 + p] * bb;
  }
  for (int o = 0; o < 64; ++o)  s += G[hh * 64 + o] * b2[o];
  for (int o = 0; o < 128; ++o) s += enc_w[hh * 128 + o] * f1b[o];
  bias_h[hh] = s;
}

// M2t[a][k] = sum_e actor_W[k,e] * emb[a,e]   (a padded to 112, pad rows = 0)
__global__ void pk5_M2t(const float* __restrict__ actor_W, const float* __restrict__ emb,
                        float* __restrict__ M2t) {
  int t = blockIdx.x * blockDim.x + threadIdx.x;
  if (t >= AP_ * 512) return;
  int a = t / 512, k = t % 512;
  float s = 0.f;
  if (a < 100)
    for (int e = 0; e < 16; ++e) s += actor_W[k * 16 + e] * emb[a * 16 + e];
  M2t[a * 512 + k] = s;
}

// Pack a row-major [O,K] f32 weight (out,in) into f16 WMMA B-fragments.
// Fragment (ot,kt): lane l, slot e holds B[k][n] = W[n_col][k] with
// n_col = ot*16 + (l&15), k = kt*32 + (e>>3)*16 + (l>>4)*8 + (e&7)   (ISA 7.12.2 B layout, wave32)
__global__ void pack_b(const float* __restrict__ src, h16* __restrict__ dst,
                       int O, int K, int KT) {
  int t = blockIdx.x * blockDim.x + threadIdx.x;
  if (t >= O * K) return;
  int e = t & 15, l = (t >> 4) & 31, fg = t >> 9;
  int kt = fg % KT, ot = fg / KT;
  int o = ot * 16 + (l & 15);
  int k = kt * 32 + ((e >> 3) & 1) * 16 + (l >> 4) * 8 + (e & 7);
  dst[t] = (h16)src[o * K + k];
}

// ---------------- encoder: scatter + folded-linear gather ----------------
// hidden[b] = bias_h + sum over winning tokens val/max_vec[atr] * T[idx,:]
// Written directly in WMMA A-fragment f16 layout.
__global__ void __launch_bounds__(128) encode_kernel(
    const int* __restrict__ obs, const float* __restrict__ max_vec,
    const float* __restrict__ T, const float* __restrict__ bias_h,
    h16* __restrict__ Apack) {
  __shared__ int   box[BOX];
  __shared__ float wv[M_];
  __shared__ int   widx[M_];
  int b = blockIdx.x, tid = threadIdx.x;

  for (int i = tid; i < BOX; i += 128) box[i] = -1;
  __syncthreads();

  const int* ob = obs + (long)b * M_ * 3;
  for (int t = tid; t < M_; t += 128) {
    int o0 = ob[t * 3 + 0]; if (o0 == 255) o0 = 0;
    int o1 = ob[t * 3 + 1]; if (o1 == 255) o1 = 0;
    int o2 = ob[t * 3 + 2]; if (o2 == 255) o2 = 0;
    int xi = (o0 >> 4) & 15, yi = o0 & 15;
    bool valid = (xi < W_) && (yi < H_) && (o1 < L_);
    int atr = valid ? o1 : 0;
    int idx = valid ? (atr * 121 + xi * 11 + yi) : -1;
    widx[t] = idx;
    wv[t]   = valid ? ((float)o2 / max_vec[atr]) : 0.f;
    if (valid) box[idx] = t;   // .set semantics: one writer wins
  }
  __syncthreads();

  for (int t = tid; t < M_; t += 128) {
    int idx = widx[t];
    bool win = (idx >= 0) && (box[idx] == t);  // dedupe: only the winner contributes
    if (!win) { wv[t] = 0.f; widx[t] = 0; }
  }
  __syncthreads();

  int hh = tid;                      // 128 threads = 128 hidden dims
  float acc = bias_h[hh];
  for (int t = 0; t < M_; ++t)
    acc += wv[t] * T[widx[t] * 128 + hh];   // coalesced over hh

  // place into A-fragment layout: m = b&15, kk = hh within K-chunk
  int bt = b >> 4, m = b & 15;
  int kt = hh >> 5, kk = hh & 31;
  int lane = m + ((kk >> 3) & 1) * 16;
  int slot = ((kk >> 4) & 1) * 8 + (kk & 7);
  Apack[(((long)bt * 4 + kt) * 32 + lane) * 16 + slot] = (h16)acc;
}

// ---------------- decoder: WMMA GEMM chain ----------------
// Per wave: 16 batch rows. critic(1024)+tanh+value dot, actor(512) -> LDS f16,
// logits = af @ M2 (K=512, N=112 padded).
__global__ void __launch_bounds__(128) decode_kernel(
    const h16* __restrict__ Apack, const h16* __restrict__ B1p,
    const h16* __restrict__ B2p, const h16* __restrict__ B3p,
    const float* __restrict__ critic1_b, const float* __restrict__ value_w,
    const float* __restrict__ value_b, const float* __restrict__ actor1_b,
    const float* __restrict__ actor_bias, float* __restrict__ out, int B) {
  extern __shared__ h16 afLDS[];           // [4 waves][16][512]
  int lane = threadIdx.x & 31;
  int wvid = threadIdx.x >> 5;
  int gw = blockIdx.x * 4 + wvid;          // global wave id = row-tile id
  int r0 = gw * 16;
  int nlo = lane & 15, hi = lane >> 4;

  v16h Afr[4];
#pragma unroll
  for (int kt = 0; kt < 4; ++kt)
    Afr[kt] = *(const v16h*)(Apack + (((long)gw * 4 + kt) * 32 + lane) * 16);

  // ---- GEMM1: crit = tanh(hidden @ W1^T + b1c); value = crit . value_w ----
  float vpart[8];
#pragma unroll
  for (int j = 0; j < 8; ++j) vpart[j] = 0.f;
  for (int ot = 0; ot < 64; ++ot) {
    v8f acc = {};
#pragma unroll
    for (int kt = 0; kt < 4; ++kt) {
      v16h Bf = *(const v16h*)(B1p + (((long)ot * 4 + kt) * 32 + lane) * 16);
      acc = __builtin_amdgcn_wmma_f32_16x16x32_f16(false, Afr[kt], false, Bf,
                                                   (short)0, acc, false, false);
    }
    int o = ot * 16 + nlo;
    float cb = critic1_b[o], vw = value_w[o];
#pragma unroll
    for (int j = 0; j < 8; ++j)
      vpart[j] += fast_tanh(acc[j] + cb) * vw;
  }
#pragma unroll
  for (int j = 0; j < 8; ++j) {   // reduce over the 16 N-lanes of each half
    float v = vpart[j];
    v += __shfl_xor(v, 1, 32);
    v += __shfl_xor(v, 2, 32);
    v += __shfl_xor(v, 4, 32);
    v += __shfl_xor(v, 8, 32);
    vpart[j] = v;
  }
  if (nlo == 0) {
    float vb = value_b[0];
#pragma unroll
    for (int j = 0; j < 8; ++j) {
      int m = j + 8 * hi;
      out[(long)B * 100 + r0 + m] = vpart[j] + vb;
    }
  }

  // ---- GEMM2: af = hidden @ W2^T + b2a  -> LDS (f16) ----
  h16* af = afLDS + wvid * 16 * 512;
  for (int ot = 0; ot < 32; ++ot) {
    v8f acc = {};
#pragma unroll
    for (int kt = 0; kt < 4; ++kt) {
      v16h Bf = *(const v16h*)(B2p + (((long)ot * 4 + kt) * 32 + lane) * 16);
      acc = __builtin_amdgcn_wmma_f32_16x16x32_f16(false, Afr[kt], false, Bf,
                                                   (short)0, acc, false, false);
    }
    int o = ot * 16 + nlo;
    float ab = actor1_b[o];
#pragma unroll
    for (int j = 0; j < 8; ++j) {
      int m = j + 8 * hi;
      af[m * 512 + o] = (h16)(acc[j] + ab);
    }
  }
  __syncthreads();

  // ---- GEMM3: logits = af @ M2 (+bias), K=512, N=112 (7 tiles) ----
  v8f lacc[7] = {};
  for (int kt = 0; kt < 16; ++kt) {
    v16h Aa = {};
#pragma unroll
    for (int e = 0; e < 16; ++e) {
      int kk = ((e >= 8) ? 16 : 0) + 8 * hi + (e & 7);   // A-frag slot -> K
      Aa[e] = af[nlo * 512 + kt * 32 + kk];
    }
#pragma unroll
    for (int nt = 0; nt < 7; ++nt) {
      v16h Bf = *(const v16h*)(B3p + (((long)nt * 16 + kt) * 32 + lane) * 16);
      lacc[nt] = __builtin_amdgcn_wmma_f32_16x16x32_f16(false, Aa, false, Bf,
                                                        (short)0, lacc[nt], false, false);
    }
  }
  float abias = actor_bias[0];
  for (int nt = 0; nt < 7; ++nt) {
    int a = nt * 16 + nlo;
    if (a < 100) {
#pragma unroll
      for (int j = 0; j < 8; ++j) {
        int m = j + 8 * hi;
        out[(long)(r0 + m) * 100 + a] = lacc[nt][j] + abias;
      }
    }
  }
}

// ---------------- launcher ----------------
extern "C" void kernel_launch(void* const* d_in, const int* in_sizes, int n_in,
                              void* d_out, int out_size, void* d_ws, size_t ws_size,
                              hipStream_t stream) {
  (void)n_in; (void)out_size; (void)ws_size;
  const int*   obs       = (const int*)d_in[0];
  const float* max_vec   = (const float*)d_in[1];
  const float* w1        = (const float*)d_in[2];
  const float* b1        = (const float*)d_in[3];
  const float* w2        = (const float*)d_in[4];
  const float* b2        = (const float*)d_in[5];
  const float* fc1_w     = (const float*)d_in[6];
  const float* f1b       = (const float*)d_in[7];
  const float* enc_w     = (const float*)d_in[8];
  const float* eb        = (const float*)d_in[9];
  const float* critic1_w = (const float*)d_in[10];
  const float* critic1_b = (const float*)d_in[11];
  const float* value_w   = (const float*)d_in[12];
  const float* value_b   = (const float*)d_in[13];
  const float* actor1_w  = (const float*)d_in[14];
  const float* actor1_b  = (const float*)d_in[15];
  const float* emb       = (const float*)d_in[16];
  const float* actor_W   = (const float*)d_in[17];
  const float* actor_bias= (const float*)d_in[18];
  float* out = (float*)d_out;
  int B = in_sizes[0] / (M_ * 3);

  char* ws = (char*)d_ws;
  size_t off = 0;
  auto alloc = [&](size_t bytes) -> void* {
    void* p = ws + off;
    off = (off + bytes + 255) & ~(size_t)255;
    return p;
  };
  float* G      = (float*)alloc(128 * 64 * 4);
  float* K2     = (float*)alloc(128 * 576 * 4);
  float* T      = (float*)alloc((size_t)BOX * 128 * 4);
  float* bias_h = (float*)alloc(128 * 4);
  float* M2t    = (float*)alloc((size_t)AP_ * 512 * 4);
  h16*   B1p    = (h16*)alloc((size_t)1024 * 128 * 2);
  h16*   B2p    = (h16*)alloc((size_t)512 * 128 * 2);
  h16*   B3p    = (h16*)alloc((size_t)AP_ * 512 * 2);
  h16*   Apack  = (h16*)alloc((size_t)B * 128 * 2);

  pk1_G  <<<(128 * 64 + 255) / 256, 256, 0, stream>>>(enc_w, fc1_w, G);
  pk2_K2 <<<(128 * 576 + 255) / 256, 256, 0, stream>>>(G, w2, K2);
  pk3_T  <<<(BOX * 128 + 255) / 256, 256, 0, stream>>>(K2, w1, T);
  pk4_bias<<<1, 128, 0, stream>>>(K2, G, enc_w, b1, b2, f1b, eb, bias_h);
  pk5_M2t<<<(AP_ * 512 + 255) / 256, 256, 0, stream>>>(actor_W, emb, M2t);
  pack_b <<<(1024 * 128 + 255) / 256, 256, 0, stream>>>(critic1_w, B1p, 1024, 128, 4);
  pack_b <<<(512 * 128 + 255) / 256, 256, 0, stream>>>(actor1_w, B2p, 512, 128, 4);
  pack_b <<<(AP_ * 512 + 255) / 256, 256, 0, stream>>>(M2t, B3p, AP_, 512, 16);

  encode_kernel<<<B, 128, 0, stream>>>(obs, max_vec, T, bias_h, Apack);
  decode_kernel<<<B / 64, 128, 4 * 16 * 512 * sizeof(h16), stream>>>(
      Apack, B1p, B2p, B3p, critic1_b, value_w, value_b, actor1_b, actor_bias, out, B);
}
